// VideoVisionMamba_44736379355747
// MI455X (gfx1250) — matrix operand order repaired and checked by reading, not compiled, not run on hardware
//
#include <hip/hip_runtime.h>

#define EMBED 768
#define DIN   1536
#define NST   16
#define DCONV 4
#define DEPTH 4
#define MLPH  3072
#define XPROJD (DIN + 2*NST)   // 1568
#define SEQ   785              // 1 + 4*14*14
#define BATCH 2
#define NTOK  (BATCH*SEQ)      // 1570
#define MR    1664             // padded token rows (13 * 128)

typedef __attribute__((ext_vector_type(16))) _Float16 v16h;
typedef __attribute__((ext_vector_type(8)))  float    v8f;

// ---------------------------------------------------------------------------
// f32 -> f16 fragment conversion (4x float4 -> 16 halfs)
// ---------------------------------------------------------------------------
__device__ __forceinline__ v16h cvt16(const float4 a0, const float4 a1,
                                      const float4 a2, const float4 a3) {
  v16h r;
  r[0]=(_Float16)a0.x;  r[1]=(_Float16)a0.y;  r[2]=(_Float16)a0.z;  r[3]=(_Float16)a0.w;
  r[4]=(_Float16)a1.x;  r[5]=(_Float16)a1.y;  r[6]=(_Float16)a1.z;  r[7]=(_Float16)a1.w;
  r[8]=(_Float16)a2.x;  r[9]=(_Float16)a2.y;  r[10]=(_Float16)a2.z; r[11]=(_Float16)a2.w;
  r[12]=(_Float16)a3.x; r[13]=(_Float16)a3.y; r[14]=(_Float16)a3.z; r[15]=(_Float16)a3.w;
  return r;
}

// A fragment (16x32): lane&15 -> M row, lane>>4 -> K half
//   elems 0..7 = K[k0+half*8 .. +7], elems 8..15 = K[k0+16+half*8 .. +7]
__device__ __forceinline__ v16h load_a_frag(const float* __restrict__ row, int k0, int khalf) {
  const float* p = row + k0 + khalf * 8;
  return cvt16(*(const float4*)(p),      *(const float4*)(p + 4),
               *(const float4*)(p + 16), *(const float4*)(p + 20));
}

// B fragment (32x16): lane&15 -> N col, elems 0..7 = K[k0+half*8 .. +7], elems 8..15 = K[k0+16+half*8 .. +7]
__device__ __forceinline__ v16h load_b_frag(const float* __restrict__ row, int k0, int khalf) {
  const float* p = row + k0 + khalf * 8;
  return cvt16(*(const float4*)(p),      *(const float4*)(p + 4),
               *(const float4*)(p + 16), *(const float4*)(p + 20));
}

// ---------------------------------------------------------------------------
// WMMA GEMM: C[M,N] = act(A[M,K] * W[N,K]^T + bias) (+ resid)
// Block = 256 threads = 8 waves arranged 4(M) x 2(N); each wave owns a 32x32
// macro-tile (2x2 WMMA tiles, 4 accumulators) -> 4 WMMAs per 8KB of operand
// loads. Block tile = 128(M) x 64(N). Rows padded to MR; N-edge W loads are
// clamped (stores masked).
// ---------------------------------------------------------------------------
__global__ __launch_bounds__(256) void gemm_wmma_f16(
    const float* __restrict__ A, const float* __restrict__ W,
    const float* __restrict__ bias, const float* __restrict__ resid,
    float* __restrict__ C, int M, int N, int K, int act)
{
  const int lane  = threadIdx.x & 31;
  const int wave  = threadIdx.x >> 5;
  const int khalf = lane >> 4;
  const int l16   = lane & 15;
  const int wm    = wave & 3;        // 4 waves across M
  const int wn    = wave >> 2;       // 2 waves across N
  const int m0 = blockIdx.y * 128 + wm * 32;
  const int n0 = blockIdx.x * 64  + wn * 32;

  const float* Arow0 = A + (size_t)(m0 + l16) * K;
  const float* Arow1 = A + (size_t)(m0 + 16 + l16) * K;
  int nc0 = n0 + l16;      if (nc0 > N - 1) nc0 = N - 1;
  int nc1 = n0 + 16 + l16; if (nc1 > N - 1) nc1 = N - 1;
  const float* Wrow0 = W + (size_t)nc0 * K;
  const float* Wrow1 = W + (size_t)nc1 * K;

  v8f c00 = {0.f,0.f,0.f,0.f,0.f,0.f,0.f,0.f};
  v8f c01 = {0.f,0.f,0.f,0.f,0.f,0.f,0.f,0.f};
  v8f c10 = {0.f,0.f,0.f,0.f,0.f,0.f,0.f,0.f};
  v8f c11 = {0.f,0.f,0.f,0.f,0.f,0.f,0.f,0.f};

#pragma unroll 2
  for (int k0 = 0; k0 < K; k0 += 32) {
    v16h a0 = load_a_frag(Arow0, k0, khalf);
    v16h a1 = load_a_frag(Arow1, k0, khalf);
    v16h b0 = load_b_frag(Wrow0, k0, khalf);
    v16h b1 = load_b_frag(Wrow1, k0, khalf);
    if (k0 + 64 < K) {
      __builtin_prefetch(Wrow0 + k0 + 64, 0, 1);   // global_prefetch_b8
      __builtin_prefetch(Arow0 + k0 + 64, 0, 1);
    }
    c00 = __builtin_amdgcn_wmma_f32_16x16x32_f16(false, a0, false, b0, (short)0, c00, false, false);
    c01 = __builtin_amdgcn_wmma_f32_16x16x32_f16(false, a0, false, b1, (short)0, c01, false, false);
    c10 = __builtin_amdgcn_wmma_f32_16x16x32_f16(false, a1, false, b0, (short)0, c10, false, false);
    c11 = __builtin_amdgcn_wmma_f32_16x16x32_f16(false, a1, false, b1, (short)0, c11, false, false);
  }

  // D layout: VGPR r, lanes 0-15 -> (mt+r, nt+lane); lanes 16-31 -> (mt+8+r, ...)
  auto store_tile = [&](const v8f& c, int mt, int nt) {
    const int col = n0 + nt * 16 + l16;
    if (col >= N) return;
    const float bv = bias ? bias[col] : 0.0f;
    const int rb = m0 + mt * 16 + (khalf ? 8 : 0);
#pragma unroll
    for (int r = 0; r < 8; ++r) {
      int row = rb + r;
      if (row < M) {
        float v = c[r] + bv;
        if (act == 1) v = 0.5f * v * (1.0f + erff(v * 0.70710678118f));  // exact GELU
        if (resid) v += resid[(size_t)row * N + col];
        C[(size_t)row * N + col] = v;
      }
    }
  };
  store_tile(c00, 0, 0);
  store_tile(c01, 0, 1);
  store_tile(c10, 1, 0);
  store_tile(c11, 1, 1);
}

// ---------------------------------------------------------------------------
// im2col of x[2,3,8,224,224] for patch conv (2,16,16 stride/kernel):
// A[r, k], r = b*784 + t*196 + h*14 + w ; k = ((c*2+kt)*16+kh)*16+kw
// ---------------------------------------------------------------------------
__global__ void im2col_kernel(const float* __restrict__ x, float* __restrict__ A, int n)
{
  int idx = blockIdx.x * 256 + threadIdx.x;
  if (idx >= n) return;
  int k = idx % 1536;
  int r = idx / 1536;
  int b = r / 784; int rem = r % 784;
  int t = rem / 196; int hh = (rem % 196) / 14; int wi = rem % 14;
  int cc = k / 512; int k2 = k % 512;
  int kt = k2 / 256; int k3 = k2 % 256;
  int kh = k3 / 16;  int kw = k3 % 16;
  size_t src = ((((size_t)b * 3 + cc) * 8 + (2 * t + kt)) * 224 + (16 * hh + kh)) * 224
               + (16 * wi + kw);
  A[(size_t)r * 1536 + k] = x[src];
}

// ---------------------------------------------------------------------------
// Assemble token stream: cls token + patch embeddings + sincos pos-embed
// ---------------------------------------------------------------------------
__global__ void assemble_kernel(const float* __restrict__ patch, const float* __restrict__ cls,
                                float* __restrict__ X, int n)
{
  int idx = blockIdx.x * 256 + threadIdx.x;
  if (idx >= n) return;
  int e = idx % EMBED;
  int r = idx / EMBED;
  int b = r / SEQ; int l = r % SEQ;
  float base = (l == 0) ? cls[e]
                        : patch[((size_t)(b * 784 + (l - 1))) * EMBED + e];
  float freq = __expf(-(float)(2 * (e >> 1)) * (9.210340371976184f / (float)EMBED));
  float ang  = (float)l * freq;
  float pe   = (e & 1) ? __cosf(ang) : __sinf(ang);
  X[(size_t)r * EMBED + e] = base + pe;
}

// ---------------------------------------------------------------------------
// LayerNorm over last dim (768), one row per 256-thread block
// ---------------------------------------------------------------------------
__global__ __launch_bounds__(256) void ln_kernel(const float* __restrict__ X,
    const float* __restrict__ w, const float* __restrict__ b,
    float* __restrict__ O, int M)
{
  __shared__ float sm[8];
  __shared__ float stat[2];
  const int row = blockIdx.x;
  if (row >= M) return;
  const float* xr = X + (size_t)row * EMBED;
  float s = 0.f;
  for (int i = threadIdx.x; i < EMBED; i += 256) s += xr[i];
  for (int off = 16; off > 0; off >>= 1) s += __shfl_xor(s, off, 32);
  if ((threadIdx.x & 31) == 0) sm[threadIdx.x >> 5] = s;
  __syncthreads();
  if (threadIdx.x == 0) {
    float t = 0.f;
    for (int i = 0; i < 8; ++i) t += sm[i];
    stat[0] = t / (float)EMBED;
  }
  __syncthreads();
  const float mean = stat[0];
  float v = 0.f;
  for (int i = threadIdx.x; i < EMBED; i += 256) { float d = xr[i] - mean; v += d * d; }
  for (int off = 16; off > 0; off >>= 1) v += __shfl_xor(v, off, 32);
  if ((threadIdx.x & 31) == 0) sm[threadIdx.x >> 5] = v;
  __syncthreads();
  if (threadIdx.x == 0) {
    float t = 0.f;
    for (int i = 0; i < 8; ++i) t += sm[i];
    stat[1] = rsqrtf(t / (float)EMBED + 1e-5f);
  }
  __syncthreads();
  const float rstd = stat[1];
  float* o = O + (size_t)row * EMBED;
  for (int i = threadIdx.x; i < EMBED; i += 256)
    o[i] = (xr[i] - mean) * rstd * w[i] + b[i];
}

// ---------------------------------------------------------------------------
// Causal depthwise conv (DCONV=4) over sequence + SiLU, from xr[..., :DIN]
// ---------------------------------------------------------------------------
__global__ void dwconv_silu_kernel(const float* __restrict__ xr, const float* __restrict__ cw,
    const float* __restrict__ cb, float* __restrict__ xs, int n)
{
  int idx = blockIdx.x * 256 + threadIdx.x;
  if (idx >= n) return;
  int d = idx % DIN;
  int r = idx / DIN;          // r = b*SEQ + l
  int l = r % SEQ;
  float acc = cb[d];
#pragma unroll
  for (int j = 0; j < DCONV; ++j) {
    int ll = l - 3 + j;
    if (ll >= 0) acc += cw[d * DCONV + j] * xr[(size_t)(r - 3 + j) * (2 * DIN) + d];
  }
  xs[(size_t)r * DIN + d] = acc / (1.0f + __expf(-acc));
}

// ---------------------------------------------------------------------------
// Selective scan, software-pipelined: one half-wave (16 lanes = 16 states)
// per (b, d) channel. Loads for the next T=8 steps are issued into a ping-
// pong register set while the current 8-step recurrence runs from registers,
// hiding L2 latency behind the dependency chain.
// ---------------------------------------------------------------------------
__global__ __launch_bounds__(256) void scan_kernel(const float* __restrict__ proj,
    const float* __restrict__ xs, const float* __restrict__ A_log,
    const float* __restrict__ Dp, float* __restrict__ y)
{
  constexpr int T = 8;
  const int lane  = threadIdx.x & 31;
  const int wglob = (blockIdx.x * 256 + threadIdx.x) >> 5;  // 0..1535
  const int b     = wglob / (DIN / 2);
  const int dpair = wglob % (DIN / 2);
  const int d     = dpair * 2 + (lane >> 4);
  const int n     = lane & 15;
  const float An  = -__expf(A_log[d * NST + n]);
  const float Dd  = Dp[d];
  const size_t rowbase = (size_t)b * SEQ;

  float h = 0.f;
  float dA[T], bA[T], cA[T], xA[T];
  float dB[T], bB[T], cB[T], xB[T];

  auto loadChunk = [&](int l0, float (&dd)[T], float (&bb)[T], float (&cc)[T], float (&xx)[T]) {
#pragma unroll
    for (int j = 0; j < T; ++j) {
      int l = l0 + j;
      if (l >= SEQ) l = SEQ - 1;            // clamp: stays in-bounds, result unused
      const float* pr = proj + (rowbase + l) * XPROJD;
      dd[j] = pr[d];
      bb[j] = pr[DIN + n];
      cc[j] = pr[DIN + NST + n];
      xx[j] = xs[(rowbase + l) * DIN + d];
    }
  };

  auto compute = [&](int l0, float (&dd)[T], float (&bb)[T], float (&cc)[T], float (&xx)[T]) {
#pragma unroll
    for (int j = 0; j < T; ++j) {
      int l = l0 + j;
      if (l >= SEQ) return;
      float draw  = dd[j];
      float delta = (draw > 20.f) ? draw : __logf(1.f + __expf(draw));  // softplus
      h = __expf(delta * An) * h + (delta * xx[j]) * bb[j];
      float t = h * cc[j];
      t += __shfl_xor(t, 1, 32);
      t += __shfl_xor(t, 2, 32);
      t += __shfl_xor(t, 4, 32);
      t += __shfl_xor(t, 8, 32);
      if (n == 0) y[(rowbase + l) * DIN + d] = t + xx[j] * Dd;
    }
  };

  loadChunk(0, dA, bA, cA, xA);
  for (int l0 = 0; l0 < SEQ; l0 += 2 * T) {
    if (l0 + T < SEQ) loadChunk(l0 + T, dB, bB, cB, xB);
    compute(l0, dA, bA, cA, xA);
    if (l0 + T >= SEQ) break;
    if (l0 + 2 * T < SEQ) loadChunk(l0 + 2 * T, dA, bA, cA, xA);
    compute(l0 + T, dB, bB, cB, xB);
  }
}

// ---------------------------------------------------------------------------
// y *= silu(res), res = xr[..., DIN:]
// ---------------------------------------------------------------------------
__global__ void gate_kernel(float* __restrict__ y, const float* __restrict__ xr, int n)
{
  int idx = blockIdx.x * 256 + threadIdx.x;
  if (idx >= n) return;
  int d = idx % DIN;
  int r = idx / DIN;
  float res = xr[(size_t)r * (2 * DIN) + DIN + d];
  y[(size_t)r * DIN + d] *= res / (1.0f + __expf(-res));
}

// ---------------------------------------------------------------------------
// Classification head: out[b,c] = <feat_b, head_w[c]> + head_b[c]
// ---------------------------------------------------------------------------
__global__ void head_kernel(const float* __restrict__ ln, const float* __restrict__ hw,
    const float* __restrict__ hb, float* __restrict__ out)
{
  int lane = threadIdx.x & 31;
  int wv   = threadIdx.x >> 5;  // 0..3
  if (wv >= 4) return;
  int b = wv >> 1, c = wv & 1;
  const float* feat = ln + (size_t)(b * SEQ) * EMBED;  // token 0 of batch b
  const float* w    = hw + c * EMBED;
  float s = 0.f;
  for (int i = lane; i < EMBED; i += 32) s += feat[i] * w[i];
  for (int off = 16; off > 0; off >>= 1) s += __shfl_xor(s, off, 32);
  if (lane == 0) out[b * 2 + c] = s + hb[c];
}

// ---------------------------------------------------------------------------
extern "C" void kernel_launch(void* const* d_in, const int* in_sizes, int n_in,
                              void* d_out, int out_size, void* d_ws, size_t ws_size,
                              hipStream_t stream)
{
  (void)in_sizes; (void)n_in; (void)out_size; (void)ws_size;
  // JAX pytree leaf order (dict keys sorted): x, cls, head_b, head_w,
  // layers[0..3]{A_log,Dp,conv_b,conv_w,fc1_b,fc1_w,fc2_b,fc2_w,in_b,in_w,
  //             ln1_b,ln1_w,ln2_b,ln2_w,out_b,out_w,xproj_b,xproj_w},
  // norm_b, norm_w, patch_b, patch_w
  const float* x       = (const float*)d_in[0];
  const float* cls     = (const float*)d_in[1];
  const float* head_b  = (const float*)d_in[2];
  const float* head_w  = (const float*)d_in[3];
  const float* norm_b  = (const float*)d_in[4 + DEPTH * 18 + 0];
  const float* norm_w  = (const float*)d_in[4 + DEPTH * 18 + 1];
  const float* patch_b = (const float*)d_in[4 + DEPTH * 18 + 2];
  const float* patch_w = (const float*)d_in[4 + DEPTH * 18 + 3];

  enum { A_LOG = 0, DP, CONV_B, CONV_W, FC1_B, FC1_W, FC2_B, FC2_W, IN_B, IN_W,
         LN1_B, LN1_W, LN2_B, LN2_W, OUT_B, OUT_W, XPROJ_B, XPROJ_W };
  auto LW = [&](int layer, int which) -> const float* {
    return (const float*)d_in[4 + layer * 18 + which];
  };

  // Workspace bump allocator (256B aligned)
  char* ws = (char*)d_ws;
  size_t off = 0;
  auto alloc = [&](size_t bytes) -> void* {
    void* p = ws + off;
    off += (bytes + 255) & ~(size_t)255;
    return p;
  };
  float* X    = (float*)alloc((size_t)MR * EMBED * 4);       // token stream (residual)
  float* lnb  = (float*)alloc((size_t)MR * EMBED * 4);       // LN output
  float* xr   = (float*)alloc((size_t)MR * (2 * DIN) * 4);   // in_proj out / mlp hidden / im2col
  float* xs   = (float*)alloc((size_t)MR * DIN * 4);         // conv+silu out
  float* proj = (float*)alloc((size_t)MR * XPROJD * 4);      // xproj out / patch gemm out
  float* yb   = (float*)alloc((size_t)MR * DIN * 4);         // scan out
  float* im2col   = xr;    // pre-layer reuse
  float* patchout = proj;  // pre-layer reuse
  float* hbuf     = xr;    // post-gate reuse for MLP hidden

  dim3 blk(256);
  const int GY = (NTOK + 127) / 128;  // 13; also covers M=1568
  auto GX = [](int N) { return (N + 63) / 64; };

  // Patch embedding (im2col + WMMA GEMM), then assemble tokens (+cls +pos)
  { int n = 1568 * 1536;
    im2col_kernel<<<(n + 255) / 256, blk, 0, stream>>>(x, im2col, n); }
  gemm_wmma_f16<<<dim3(GX(EMBED), GY), blk, 0, stream>>>(
      im2col, patch_w, patch_b, nullptr, patchout, 1568, EMBED, 1536, 0);
  { int n = NTOK * EMBED;
    assemble_kernel<<<(n + 255) / 256, blk, 0, stream>>>(patchout, cls, X, n); }

  for (int d = 0; d < DEPTH; ++d) {
    ln_kernel<<<NTOK, blk, 0, stream>>>(X, LW(d, LN1_W), LW(d, LN1_B), lnb, NTOK);
    gemm_wmma_f16<<<dim3(GX(2 * DIN), GY), blk, 0, stream>>>(
        lnb, LW(d, IN_W), LW(d, IN_B), nullptr, xr, NTOK, 2 * DIN, EMBED, 0);
    { int n = NTOK * DIN;
      dwconv_silu_kernel<<<(n + 255) / 256, blk, 0, stream>>>(
          xr, LW(d, CONV_W), LW(d, CONV_B), xs, n); }
    gemm_wmma_f16<<<dim3(GX(XPROJD), GY), blk, 0, stream>>>(
        xs, LW(d, XPROJ_W), LW(d, XPROJ_B), nullptr, proj, NTOK, XPROJD, DIN, 0);
    scan_kernel<<<192, blk, 0, stream>>>(proj, xs, LW(d, A_LOG), LW(d, DP), yb);
    { int n = NTOK * DIN;
      gate_kernel<<<(n + 255) / 256, blk, 0, stream>>>(yb, xr, n); }
    gemm_wmma_f16<<<dim3(GX(EMBED), GY), blk, 0, stream>>>(
        yb, LW(d, OUT_W), LW(d, OUT_B), X, X, NTOK, EMBED, DIN, 0);
    ln_kernel<<<NTOK, blk, 0, stream>>>(X, LW(d, LN2_W), LW(d, LN2_B), lnb, NTOK);
    gemm_wmma_f16<<<dim3(GX(MLPH), GY), blk, 0, stream>>>(
        lnb, LW(d, FC1_W), LW(d, FC1_B), nullptr, hbuf, NTOK, MLPH, EMBED, 1);
    gemm_wmma_f16<<<dim3(GX(EMBED), GY), blk, 0, stream>>>(
        hbuf, LW(d, FC2_W), LW(d, FC2_B), X, X, NTOK, EMBED, MLPH, 0);
  }

  ln_kernel<<<NTOK, blk, 0, stream>>>(X, norm_w, norm_b, lnb, NTOK);
  head_kernel<<<1, 128, 0, stream>>>(lnb, head_w, head_b, (float*)d_out);
}
